// VQNeighbor2_26405458936342
// MI455X (gfx1250) — matrix operand order repaired and checked by reading, not compiled, not run on hardware
//
#include <hip/hip_runtime.h>
#include <hip/hip_bf16.h>

typedef __attribute__((ext_vector_type(16))) __bf16 v16bf;
typedef __attribute__((ext_vector_type(8)))  float  v8f;

#define B_   64
#define T_   1024
#define D_   128
#define NE_  512            // n_e
#define NC_  513            // n_e + 1 columns
#define NT_  33             // 33 n-tiles of 16 -> 528 (padded)
#define MT_  (B_ * T_)      // 65536 rows
#define LEG  0.2f

// ---- workspace layout (bytes) ----
#define OFF_CBITS  ((size_t)0)                       // MT_ * 16 u32  = 4 MB
#define OFF_MINIDX ((size_t)(OFF_CBITS  + (size_t)MT_ * 16 * 4))
#define OFF_ENC    ((size_t)(OFF_MINIDX + (size_t)MT_ * 4))
#define OFF_ENERGY ((size_t)(OFF_ENC    + (size_t)MT_ * 4))
#define OFF_SCAL   ((size_t)(OFF_ENERGY + (size_t)MT_ * 4))
#define OFF_EBH    ((size_t)(OFF_SCAL + 64))         // 513*128 bf16 (hi)   = 131328 B
#define OFF_EBL    ((size_t)(OFF_EBH + (size_t)NC_ * D_ * 2))
#define OFF_EE     ((size_t)(OFF_EBL + (size_t)NC_ * D_ * 2))   // 513 f32
#define WS_NEEDED  ((size_t)(OFF_EE + 2112))

// ---- output layout (floats) ----
#define O_KH   ((size_t)0)
#define O_ENC  ((size_t)(B_*(size_t)T_*D_))          // 8388608
#define O_V    ((size_t)(O_ENC + MT_))               // 8454144
#define O_LH   ((size_t)(O_V + 1))                   // 8454145
#define O_LN   ((size_t)(O_LH + MT_))                // 8519681
#define O_EM   ((size_t)(O_LN + MT_))                // 8585217
#define O_LED  ((size_t)(O_EM + 1))                  // 8585218

// =====================================================================
// Kernel 0: one-shot prep — split emb into bf16 hi/lo (row-major, which
// is exactly the per-lane WMMA-B fragment layout) and precompute ||e_n||^2.
// One wave per embedding row.
// =====================================================================
__global__ __launch_bounds__(32)
void vq_prep_kernel(const float* __restrict__ emb, __bf16* __restrict__ ebh,
                    __bf16* __restrict__ ebl, float* __restrict__ ee)
{
    const int n    = blockIdx.x;        // 0..512
    const int lane = threadIdx.x;
    const float4 x = ((const float4*)(emb + (size_t)n * D_))[lane];

    float s = x.x * x.x + x.y * x.y + x.z * x.z + x.w * x.w;

    const size_t o = (size_t)n * D_ + (size_t)lane * 4;
    const __bf16 h0 = (__bf16)x.x, h1 = (__bf16)x.y, h2 = (__bf16)x.z, h3 = (__bf16)x.w;
    ebh[o + 0] = h0; ebh[o + 1] = h1; ebh[o + 2] = h2; ebh[o + 3] = h3;
    ebl[o + 0] = (__bf16)(x.x - (float)h0);
    ebl[o + 1] = (__bf16)(x.y - (float)h1);
    ebl[o + 2] = (__bf16)(x.z - (float)h2);
    ebl[o + 3] = (__bf16)(x.w - (float)h3);

    #pragma unroll
    for (int off = 16; off; off >>= 1) s += __shfl_xor(s, off);
    if (lane == 0) ee[n] = s;
}

// =====================================================================
// Kernel 1: split-bf16 WMMA distance GEMM, fused row argmin + advance-bit
// emission.  8 waves/block, each wave owns a 16-row m-tile and sweeps all
// 33 n-tiles.  B fragments are direct bf16 vector loads (L2-hot).
// =====================================================================
__global__ __launch_bounds__(256)
void vq_dist_kernel(const float* __restrict__ key, const __bf16* __restrict__ ebh,
                    const __bf16* __restrict__ ebl, const float* __restrict__ ee,
                    unsigned* __restrict__ cbits, int* __restrict__ minidx)
{
    __shared__ float tile[8][16][17];   // per-wave C-tile transpose buffer (pad 17 vs banks)

    const int lane = threadIdx.x & 31;
    const int wave = threadIdx.x >> 5;
    const int half = lane >> 4;         // 0: lanes 0-15, 1: lanes 16-31
    const int col  = lane & 15;
    const int m0   = (blockIdx.x * 8 + wave) * 16;

    // ---- A fragments: 16 rows x 128 K, bf16 hi/lo split (once per wave) ----
    // A k order used here (contiguous, same as B): lanes<16 elem j -> K=j; lanes>=16 elem j -> K=16+j.
    v16bf ahi[4], alo[4];
    {
        const float* arow = key + (size_t)(m0 + col) * D_;
        const int klane = half ? 8 : 0;
        #pragma unroll
        for (int kc = 0; kc < 4; ++kc) {
            #pragma unroll
            for (int j = 0; j < 16; ++j) {
                const int k = kc * 32 + half * 16 + j;
                const float x = arow[k];
                const __bf16 h = (__bf16)x;
                ahi[kc][j] = h;
                alo[kc][j] = (__bf16)(x - (float)h);
            }
        }
    }

    // row-scan state (lanes 0..15 each own one output row)
    float    dprev = 0.0f;
    unsigned bits  = 0u;
    float    minv  = 3.0e38f;
    int      mini  = 0;

    for (int nt = 0; nt < NT_; ++nt) {
        const int n  = nt * 16 + col;
        const int nc = (n > NE_) ? NE_ : n;                  // clamp pad cols
        const size_t base = (size_t)nc * D_ + (size_t)half * 16;

        // B fragments (32x16 per k-chunk): 16 consecutive bf16 per lane, pre-split
        v16bf bh[4], bl[4];
        #pragma unroll
        for (int kc = 0; kc < 4; ++kc) {
            bh[kc] = *(const v16bf*)(ebh + base + (size_t)kc * 32);
            bl[kc] = *(const v16bf*)(ebl + base + (size_t)kc * 32);
        }
        const float een = (n > NE_) ? 3.0e38f : ee[nc];      // pad columns never win

        // acc = sum_k (Alo*Bhi + Ahi*Blo + Ahi*Bhi), f32 accumulate
        v8f acc = {};
        #pragma unroll
        for (int kc = 0; kc < 4; ++kc) {
            acc = __builtin_amdgcn_wmma_f32_16x16x32_bf16(false, alo[kc], false, bh[kc],
                                                          (short)0, acc, false, false);
            acc = __builtin_amdgcn_wmma_f32_16x16x32_bf16(false, ahi[kc], false, bl[kc],
                                                          (short)0, acc, false, false);
            acc = __builtin_amdgcn_wmma_f32_16x16x32_bf16(false, ahi[kc], false, bh[kc],
                                                          (short)0, acc, false, false);
        }

        // s = ||e||^2 - 2*k.e  (||k||^2 dropped: constant per row) -> LDS transpose
        #pragma unroll
        for (int v = 0; v < 8; ++v)
            tile[wave][v + 8 * half][col] = een - 2.0f * acc[v];

        asm volatile("s_wait_dscnt 0" ::: "memory");   // wave-private buffer, no barrier

        if (lane < 16) {                   // lane r scans row r of the tile
            #pragma unroll
            for (int c = 0; c < 16; ++c) {
                const int   ng = nt * 16 + c;
                const float d  = tile[wave][lane][c];
                if (ng > 0 && ng <= 511 && d < dprev)      // advance-bit: d[n] < d[n-1]
                    bits |= 1u << ((ng - 1) & 31);
                if (ng > 0 && ng <= 512 && (ng & 31) == 0) {
                    cbits[(size_t)(m0 + lane) * 16 + (size_t)((ng - 1) >> 5)] = bits;
                    bits = 0u;
                }
                if (ng < NC_ && d < minv) { minv = d; mini = ng; }   // first-min tiebreak
                dprev = d;
            }
        }
    }
    if (lane < 16) minidx[m0 + lane] = mini;
}

// =====================================================================
// Kernel 2: sequential neighbor scan, one wave per batch b.
// Bit-row loads are coalesced & independent; only the ALU chain is serial.
// =====================================================================
__global__ __launch_bounds__(32)
void vq_scan_kernel(const unsigned* __restrict__ cbits, const int* __restrict__ minidx,
                    const unsigned char* __restrict__ zmask,
                    int* __restrict__ enc, int* __restrict__ vmax)
{
    const int b = blockIdx.x, lane = threadIdx.x;
    int ind = minidx[(size_t)b * T_];
    if (ind > NE_ - 1) ind = NE_ - 1;       // clip(argmin, 0, 511)
    if (zmask[b]) ind = 0;
    if (lane == 0) enc[(size_t)b * T_] = ind;
    const int ind0 = ind;

    for (int t = 1; t < T_; ++t) {
        const unsigned* row = cbits + ((size_t)b * T_ + t) * 16;
        const unsigned  w   = row[lane & 15];            // lanes 0..15 hold the 16 words
        const unsigned  sel = __shfl(w, ind >> 5);
        ind += (sel >> (ind & 31)) & 1u;                 // bit 511 is 0 -> saturates
        if (lane == 0) enc[(size_t)b * T_ + t] = ind;
    }
    if (lane == 0) atomicMax(vmax, ind - ind0);          // enc monotone: max-min = last-first
}

// =====================================================================
// Kernel 3: gather epilogue, one wave per (b,t). Exact f32 losses.
// =====================================================================
__global__ __launch_bounds__(256)
void vq_epilogue_kernel(const float* __restrict__ key, const float* __restrict__ emb,
                        const int* __restrict__ enc, const int* __restrict__ minidx,
                        float* __restrict__ energy, float* __restrict__ out_kh,
                        float* __restrict__ out_enc, float* __restrict__ out_lh,
                        float* __restrict__ out_ln)
{
    const int gw   = blockIdx.x * 8 + (threadIdx.x >> 5);   // 0..65535
    const int lane = threadIdx.x & 31;

    const int eh = enc[gw];
    int       en = eh + 1; if (en > NE_ - 1) en = NE_ - 1;
    const int em = minidx[gw];                               // may be 512 (extra row)

    const float4 k = ((const float4*)(key + (size_t)gw * D_))[lane];
    const float4 h = ((const float4*)(emb + (size_t)eh * D_))[lane];
    const float4 n = ((const float4*)(emb + (size_t)en * D_))[lane];
    const float4 m = ((const float4*)(emb + (size_t)em * D_))[lane];

    float4 kh;                                               // straight-through forward
    kh.x = k.x + (h.x - k.x); kh.y = k.y + (h.y - k.y);
    kh.z = k.z + (h.z - k.z); kh.w = k.w + (h.w - k.w);
    ((float4*)(out_kh + (size_t)gw * D_))[lane] = kh;

    float sh = (k.x-h.x)*(k.x-h.x) + (k.y-h.y)*(k.y-h.y) + (k.z-h.z)*(k.z-h.z) + (k.w-h.w)*(k.w-h.w);
    float sn = (k.x-n.x)*(k.x-n.x) + (k.y-n.y)*(k.y-n.y) + (k.z-n.z)*(k.z-n.z) + (k.w-n.w)*(k.w-n.w);
    float sm = (k.x-m.x)*(k.x-m.x) + (k.y-m.y)*(k.y-m.y) + (k.z-m.z)*(k.z-m.z) + (k.w-m.w)*(k.w-m.w);
    #pragma unroll
    for (int off = 16; off; off >>= 1) {
        sh += __shfl_xor(sh, off); sn += __shfl_xor(sn, off); sm += __shfl_xor(sm, off);
    }

    if (lane == 0) {
        const float lhb = sh + LEG * sh;     // vq_loss forward = (1+LEGACY)*sum
        const float lnb = sn + LEG * sn;
        const float lmv = sm + LEG * sm;
        const float dif = sn - sh;
        energy[gw]  = dif + LEG * dif;
        out_enc[gw] = (float)eh;
        out_lh[gw]  = lhb - lnb - ((lmv < lhb) ? lmv : 0.0f);
        out_ln[gw]  = lnb - lhb - ((lmv < lnb) ? lmv : 0.0f);
    }
}

// =====================================================================
// Kernel 4/5/6: scalar reductions
// =====================================================================
__global__ void vq_init_kernel(float* sums, int* vmax)
{
    if (threadIdx.x == 0) { sums[0] = 0.0f; sums[1] = 0.0f; *vmax = 0; }
}

__global__ __launch_bounds__(256)
void vq_reduce_kernel(const float* __restrict__ energy, const int* __restrict__ enc,
                      float* __restrict__ sums)
{
    __shared__ float sE[8], sL[8];
    float e_sum = 0.0f, l_sum = 0.0f;
    for (int i = blockIdx.x * 256 + threadIdx.x; i < MT_; i += gridDim.x * 256) {
        const float e = energy[i];
        e_sum += e;
        if ((i & (T_ - 1)) > 0) {
            const float diff = (enc[i] == enc[i - 1]) ? (e - energy[i - 1]) : 0.0f;
            const float v = diff + (1e-6f / (float)NE_);
            l_sum += (v > 0.0f) ? v : 0.0f;
        }
    }
    #pragma unroll
    for (int off = 16; off; off >>= 1) {
        e_sum += __shfl_xor(e_sum, off); l_sum += __shfl_xor(l_sum, off);
    }
    const int lane = threadIdx.x & 31, wv = threadIdx.x >> 5;
    if (lane == 0) { sE[wv] = e_sum; sL[wv] = l_sum; }
    __syncthreads();
    if (threadIdx.x == 0) {
        float te = 0.0f, tl = 0.0f;
        for (int i = 0; i < 8; ++i) { te += sE[i]; tl += sL[i]; }
        atomicAdd(&sums[0], te);
        atomicAdd(&sums[1], tl);
    }
}

__global__ void vq_finalize_kernel(const float* sums, const int* vmax,
                                   float* out_v, float* out_em, float* out_led)
{
    if (threadIdx.x == 0) {
        *out_v   = (float)(*vmax);
        *out_em  = sums[0] / (float)MT_;
        *out_led = sums[1] / (float)(B_ * (T_ - 1));
    }
}

extern "C" void kernel_launch(void* const* d_in, const int* in_sizes, int n_in,
                              void* d_out, int out_size, void* d_ws, size_t ws_size,
                              hipStream_t stream)
{
    if (ws_size < WS_NEEDED) return;

    const float*         key   = (const float*)d_in[0];
    const float*         emb   = (const float*)d_in[1];
    const unsigned char* zmask = (const unsigned char*)d_in[2];

    char*     ws     = (char*)d_ws;
    unsigned* cbits  = (unsigned*)(ws + OFF_CBITS);
    int*      minidx = (int*)(ws + OFF_MINIDX);
    int*      enc    = (int*)(ws + OFF_ENC);
    float*    energy = (float*)(ws + OFF_ENERGY);
    float*    sums   = (float*)(ws + OFF_SCAL);
    int*      vmax   = (int*)(ws + OFF_SCAL + 8);
    __bf16*   ebh    = (__bf16*)(ws + OFF_EBH);
    __bf16*   ebl    = (__bf16*)(ws + OFF_EBL);
    float*    ee     = (float*)(ws + OFF_EE);

    float* out = (float*)d_out;

    vq_init_kernel<<<dim3(1), dim3(32), 0, stream>>>(sums, vmax);
    vq_prep_kernel<<<dim3(NC_), dim3(32), 0, stream>>>(emb, ebh, ebl, ee);
    vq_dist_kernel<<<dim3(MT_ / 128), dim3(256), 0, stream>>>(key, ebh, ebl, ee, cbits, minidx);
    vq_scan_kernel<<<dim3(B_), dim3(32), 0, stream>>>(cbits, minidx, zmask, enc, vmax);
    vq_epilogue_kernel<<<dim3(MT_ / 8), dim3(256), 0, stream>>>(key, emb, enc, minidx, energy,
                                                                out + O_KH, out + O_ENC,
                                                                out + O_LH, out + O_LN);
    vq_reduce_kernel<<<dim3(64), dim3(256), 0, stream>>>(energy, enc, sums);
    vq_finalize_kernel<<<dim3(1), dim3(32), 0, stream>>>(sums, vmax,
                                                         out + O_V, out + O_EM, out + O_LED);
}
